// TCBlock_86878598463552
// MI455X (gfx1250) — matrix operand order repaired and checked by reading, not compiled, not run on hardware
//
#include <hip/hip_runtime.h>

typedef float    v2f  __attribute__((ext_vector_type(2)));
typedef float    v8f  __attribute__((ext_vector_type(8)));
typedef _Float16 v8h  __attribute__((ext_vector_type(8)));
typedef _Float16 v16h __attribute__((ext_vector_type(16)));

#define NMODES   128
#define NCH      256
#define NBATCH   32
#define NLEN     4096
#define NROWS    (NBATCH * NCH)   // 8192
#define TWO_PI_F 6.283185307179586f

// ---------------------------------------------------------------------------
// WMMA wrappers
// ---------------------------------------------------------------------------
__device__ __forceinline__ v8f wmma_f32x4(v2f a, v2f b, v8f c) {
    return __builtin_amdgcn_wmma_f32_16x16x4_f32(
        false, a, false, b, (short)0, c, false, false);
}
__device__ __forceinline__ v8f wmma_f16x32(v16h a, v16h b, v8f c) {
    return __builtin_amdgcn_wmma_f32_16x16x32_f16(
        false, a, false, b, (short)0, c, false, false);
}
// Two aligned 16B LDS runs -> one v16h fragment (2x ds_load_b128)
__device__ __forceinline__ v16h ld_frag(const _Float16* p0, const _Float16* p1) {
    v8h a = *(const v8h*)p0;
    v8h b = *(const v8h*)p1;
    return __builtin_shufflevector(a, b, 0,1,2,3,4,5,6,7,8,9,10,11,12,13,14,15);
}

// ---------------------------------------------------------------------------
// Basis builders (phase reduced mod 4096 so __sincosf stays accurate).
// ---------------------------------------------------------------------------
__global__ void build_fwd_basis(float* __restrict__ Bf) {
    int idx = blockIdx.x * blockDim.x + threadIdx.x;
    if (idx >= NLEN * NMODES) return;
    int n = idx >> 7;
    int m = idx & (NMODES - 1);
    int ph = (n * m) & (NLEN - 1);
    float ang = -(TWO_PI_F / (float)NLEN) * (float)ph;
    float s, c;
    __sincosf(ang, &s, &c);
    Bf[(size_t)n * 256 + m]       = c;   // Re
    Bf[(size_t)n * 256 + 128 + m] = s;   // Im (negative sin via negative angle)
}

// Inverse, UNSCALED (1/N applied in epilogue; keeps f16 lo-parts off subnormals)
__global__ void build_inv_basis(float* __restrict__ Binv) {
    int idx = blockIdx.x * blockDim.x + threadIdx.x;
    if (idx >= 256 * NLEN) return;
    int c = idx >> 12;
    int n = idx & (NLEN - 1);
    int m = c & (NMODES - 1);
    int ph = (n * m) & (NLEN - 1);
    float ang = (TWO_PI_F / (float)NLEN) * (float)ph;
    float s, co;
    __sincosf(ang, &s, &co);
    float v;
    if (c < 128) v = ((m == 0) ? 1.0f : 2.0f) * co;
    else         v = (m == 0) ? 0.0f : -2.0f * s;
    Binv[(size_t)c * NLEN + n] = v;
}

// ---------------------------------------------------------------------------
// Split-precision f16 WMMA GEMM: C = scale * (A * B), fp32-grade accuracy via
// hi/lo decomposition: ah*bh + ah*bl + al*bh per K=32 chunk (3 WMMAs/tile).
// Block tile 128x64x32, 256 threads = 8 waves (4m x 2n), wave = 2x2 subtiles.
// A staged [row][K] (stride 40 halves), B staged TRANSPOSED [col][K] so every
// fragment is two aligned b128 LDS loads, bank-conflict-free (stride 20 dw).
// FUSE=1: C = Xres + leakyrelu_{0.2}(scale * A*B)
// ---------------------------------------------------------------------------
#define BM  128
#define BN  64
#define BK  32
#define SAS 40   // A LDS row stride (halves): 16B-aligned runs, 20-dword pitch
#define SBT 40   // B^T LDS col stride (halves)

template <int FUSE>
__global__ __launch_bounds__(256)
void gemm_f16split(const float* __restrict__ A, const float* __restrict__ Bm,
                   const float* __restrict__ Xres, float* __restrict__ C,
                   int M, int N, int K, float scale) {
    __shared__ _Float16 sAhi[BM * SAS];
    __shared__ _Float16 sAlo[BM * SAS];
    __shared__ _Float16 sBhi[BN * SBT];   // transposed: [col][K]
    __shared__ _Float16 sBlo[BN * SBT];
    const int tid  = threadIdx.x;
    const int lane = tid & 31;
    const int wid  = tid >> 5;
    const int wm   = wid >> 1;               // 0..3
    const int wn   = wid & 1;                // 0..1
    const int m0   = blockIdx.y * BM;
    const int n0   = blockIdx.x * BN;

    v8f acc[2][2] = {};

    for (int k0 = 0; k0 < K; k0 += BK) {
        // Stage A tile (128x32) as hi/lo halves
        for (int i = tid; i < BM * BK; i += 256) {
            int r = i >> 5, kk = i & 31;
            float v = A[(size_t)(m0 + r) * K + (k0 + kk)];
            _Float16 h = (_Float16)v;
            sAhi[r * SAS + kk] = h;
            sAlo[r * SAS + kk] = (_Float16)(v - (float)h);
        }
        // Stage B tile (32x64) transposed as hi/lo halves
        for (int i = tid; i < BK * BN; i += 256) {
            int r = i >> 6, cc = i & 63;
            float v = Bm[(size_t)(k0 + r) * N + (n0 + cc)];
            _Float16 h = (_Float16)v;
            sBhi[cc * SBT + r] = h;
            sBlo[cc * SBT + r] = (_Float16)(v - (float)h);
        }
        if (k0 + BK < K)
            __builtin_prefetch(&A[(size_t)(m0 + (tid >> 1)) * K + (k0 + BK)], 0, 1);
        __syncthreads();

        const int lrow = lane & 15;
        const int hiA  = (lane >> 4) << 3;   // A: +8  K offset for upper half-wave
        const int hiB  = (lane >> 4) << 3;   // B: +8  K offset for upper half-wave

        v16h ah[2], al[2], bh[2], bl[2];
        #pragma unroll
        for (int s = 0; s < 2; ++s) {
            // A 16x32 f16 layout: K = (h&7) | (h>=8 ? 16:0) | (lane>=16 ? 8:0)
            const int ar = (wm * 32 + s * 16 + lrow) * SAS;
            ah[s] = ld_frag(&sAhi[ar + hiA], &sAhi[ar + 16 + hiA]);
            al[s] = ld_frag(&sAlo[ar + hiA], &sAlo[ar + 16 + hiA]);
            // B 32x16 f16 layout: K = (h&7) | (h>=8 ? 16:0) | (lane>=16 ? 8:0), N = lane&15
            const int bc = (wn * 32 + s * 16 + lrow) * SBT + hiB;
            bh[s] = ld_frag(&sBhi[bc], &sBhi[bc + 16]);
            bl[s] = ld_frag(&sBlo[bc], &sBlo[bc + 16]);
        }
        #pragma unroll
        for (int sm = 0; sm < 2; ++sm)
            #pragma unroll
            for (int sn = 0; sn < 2; ++sn) {
                acc[sm][sn] = wmma_f16x32(ah[sm], bh[sn], acc[sm][sn]);
                acc[sm][sn] = wmma_f16x32(ah[sm], bl[sn], acc[sm][sn]);
                acc[sm][sn] = wmma_f16x32(al[sm], bh[sn], acc[sm][sn]);
            }
        __syncthreads();
    }

    // Epilogue: C/D layout = 8 VGPRs; lanes<16: M=v, lanes>=16: M=8+v; N=lane&15
    const int lrow = lane & 15;
    const int roff = (lane >> 4) * 8;
    #pragma unroll
    for (int sm = 0; sm < 2; ++sm)
        #pragma unroll
        for (int sn = 0; sn < 2; ++sn)
            #pragma unroll
            for (int v = 0; v < 8; ++v) {
                const int r = m0 + wm * 32 + sm * 16 + roff + v;
                const int cidx = n0 + wn * 32 + sn * 16 + lrow;
                const size_t o = (size_t)r * N + cidx;
                float val = acc[sm][sn][v] * scale;
                if (FUSE) {
                    float act = (val >= 0.0f) ? val : 0.2f * val;
                    C[o] = Xres[o] + act;
                } else {
                    C[o] = val;
                }
            }
}

// ---------------------------------------------------------------------------
// Per-mode complex channel mix (f32 WMMA; 6% of FLOPs, full precision).
// Staging uses GLOBAL_LOAD_ASYNC_TO_LDS_B32 (ASYNCcnt-tracked, no VGPR hop):
// per-lane LDS dest = low 32 bits of flat shared address (ISA 10.2 aperture).
//   O[b,o] = sum_i (Xr + iXi)[b,i] * (Wr + iWi)[o,i]
// ---------------------------------------------------------------------------
__device__ __forceinline__ void async_g2l_b32(const float* gaddr, float* lds) {
    unsigned ldsa = (unsigned)(uintptr_t)lds;
    asm volatile("global_load_async_to_lds_b32 %0, %1, off"
                 :: "v"(ldsa), "v"(gaddr) : "memory");
}

__global__ __launch_bounds__(256)
void mode_mix(const float* __restrict__ Xft, const float* __restrict__ W,
              float* __restrict__ Oft) {
    __shared__ float sXr[NBATCH * 17];
    __shared__ float sXi[NBATCH * 17];
    __shared__ float sWr[16 * 65];
    __shared__ float sWi[16 * 65];
    const int m    = blockIdx.y;
    const int o0   = blockIdx.x * 64;
    const int tid  = threadIdx.x;
    const int lane = tid & 31;
    const int wid  = tid >> 5;
    const int wm   = wid >> 2;   // 0..1  (b halves)
    const int wn   = wid & 3;    // 0..3  (o subtiles)

    v8f accR = {};
    v8f accI = {};

    for (int i0 = 0; i0 < NCH; i0 += 16) {
        // Async-copy X tiles (32 x 16 Re/Im) and W tiles (16 x 64 Re/Im)
        for (int i = tid; i < NBATCH * 16; i += 256) {
            int b = i >> 4, ii = i & 15;
            size_t base = (size_t)(b * NCH + i0 + ii) * 256;
            async_g2l_b32(Xft + base + m,       &sXr[b * 17 + ii]);
            async_g2l_b32(Xft + base + 128 + m, &sXi[b * 17 + ii]);
        }
        for (int i = tid; i < 16 * 64; i += 256) {
            int ii = i >> 6, o = i & 63;
            size_t base = (((size_t)(o0 + o) * NCH + (i0 + ii)) * NMODES + m) * 2;
            async_g2l_b32(W + base,     &sWr[ii * 65 + o]);
            async_g2l_b32(W + base + 1, &sWi[ii * 65 + o]);
        }
        asm volatile("s_wait_asynccnt 0x0" ::: "memory");
        __syncthreads();

        const int lrow  = lane & 15;
        const int khalf = (lane >> 4) << 1;
        #pragma unroll
        for (int kk = 0; kk < 4; ++kk) {
            const int kb = kk * 4 + khalf;
            const int ar = wm * 16 + lrow;
            v2f arf, aif, brf, bif, binf;
            arf.x = sXr[ar * 17 + kb];     arf.y = sXr[ar * 17 + kb + 1];
            aif.x = sXi[ar * 17 + kb];     aif.y = sXi[ar * 17 + kb + 1];
            const int bc = wn * 16 + lrow;
            brf.x = sWr[kb * 65 + bc];     brf.y = sWr[(kb + 1) * 65 + bc];
            bif.x = sWi[kb * 65 + bc];     bif.y = sWi[(kb + 1) * 65 + bc];
            binf.x = -bif.x;               binf.y = -bif.y;
            accR = wmma_f32x4(arf, brf, accR);
            accR = wmma_f32x4(aif, binf, accR);
            accI = wmma_f32x4(arf, bif, accI);
            accI = wmma_f32x4(aif, brf, accI);
        }
        __syncthreads();
    }

    const int lrow = lane & 15;
    const int roff = (lane >> 4) * 8;
    #pragma unroll
    for (int v = 0; v < 8; ++v) {
        const int b = wm * 16 + roff + v;
        const int o = o0 + wn * 16 + lrow;
        size_t base = (size_t)(b * NCH + o) * 256;
        Oft[base + m]       = accR[v];
        Oft[base + 128 + m] = accI[v];
    }
}

// ---------------------------------------------------------------------------
// Launch: bases -> fwd DFT (f16-split GEMM) -> complex mix -> fused inv GEMM
// Workspace (floats): Bf 1M | Binv 1M | Xft 2M | Oft 2M  = 24 MB
// ---------------------------------------------------------------------------
extern "C" void kernel_launch(void* const* d_in, const int* in_sizes, int n_in,
                              void* d_out, int out_size, void* d_ws, size_t ws_size,
                              hipStream_t stream) {
    (void)in_sizes; (void)n_in; (void)out_size; (void)ws_size;
    const float* x = (const float*)d_in[0];   // (32, 256, 4096)
    const float* w = (const float*)d_in[1];   // (256, 256, 128, 2)
    float* out = (float*)d_out;               // (32, 256, 4096)

    float* ws   = (float*)d_ws;
    float* Bf   = ws;                                  // 4096 x 256
    float* Binv = Bf   + (size_t)NLEN * 256;           // 256 x 4096 (unscaled)
    float* Xft  = Binv + (size_t)256 * NLEN;           // 8192 x 256
    float* Oft  = Xft  + (size_t)NROWS * 256;          // 8192 x 256

    build_fwd_basis<<<(NLEN * NMODES) / 256, 256, 0, stream>>>(Bf);
    build_inv_basis<<<(256 * NLEN) / 256, 256, 0, stream>>>(Binv);

    // Xft[8192 x 256] = x[8192 x 4096] * Bf[4096 x 256]
    gemm_f16split<0><<<dim3(256 / BN, NROWS / BM), 256, 0, stream>>>(
        x, Bf, nullptr, Xft, NROWS, 256, NLEN, 1.0f);

    // Per-mode complex mix
    mode_mix<<<dim3(NCH / 64, NMODES), 256, 0, stream>>>(Xft, w, Oft);

    // out = x + leakyrelu((1/N) * Oft[8192 x 256] * Binv[256 x 4096])
    gemm_f16split<1><<<dim3(NLEN / BN, NROWS / BM), 256, 0, stream>>>(
        Oft, Binv, x, out, NROWS, NLEN, 256, 1.0f / (float)NLEN);
}
